// TransformerMoETaskGating_90400471646252
// MI455X (gfx1250) — compile-verified, matrix operand order checked by reading, not run on hardware
//
#include <hip/hip_runtime.h>
#include <hip/hip_bf16.h>

// ---------------- problem constants (from reference) ----------------
#define Dd    4
#define Tt    4
#define DIMv  384
#define EAv   12
#define Hh    6
#define HDv   64
#define EMv   8
#define KMv   2
#define HMv   768
#define NCLSv 2
#define Bb    8
#define Nn    384
#define TOK   (Bb * Nn)          // 3072
#define SCALEv 0.125f            // 64^-0.5
#define SWITCHv 0.01f
#define ZLOSSv  0.001f

typedef __attribute__((ext_vector_type(16))) _Float16 v16h;
typedef __attribute__((ext_vector_type(8)))  _Float16 h8v;
typedef __attribute__((ext_vector_type(8)))  float    v8f;

// =====================================================================
// Batched WMMA GEMM, 2x2 register-blocked (one wave owns a 32x32 C tile):
//   C[b][M,N] (f32, row-major) = A[b][M,K] (f16 row-major)
//                              * Bt[b/bdivB][N,K] (f16 row-major, i.e. B^T)
// Per K-step: 8 b128 loads (2 A frags + 2 B frags) feed 4 independent
// v_wmma_f32_16x16x32_f16 -> 16 FLOP per L2 byte, 4-way WMMA ILP.
// M,N multiples of 32; K multiple of 32.
// =====================================================================
static __device__ __forceinline__ v16h frag_ld(const _Float16* p, int k0) {
  h8v lo = *(const h8v*)(p + k0);
  h8v hi = *(const h8v*)(p + k0 + 16);
  return __builtin_shufflevector(lo, hi, 0,1,2,3,4,5,6,7,8,9,10,11,12,13,14,15);
}

__global__ __launch_bounds__(256) void gemm_wmma_f16(
    const _Float16* __restrict__ A, const _Float16* __restrict__ Bt,
    float* __restrict__ C, int M, int N, int K,
    long long sA, long long sB, int bdivB, long long sC, int batch)
{
  int lane = threadIdx.x & 31;
  long long wave = ((long long)blockIdx.x * blockDim.x + threadIdx.x) >> 5;
  int tilesM = M >> 5, tilesN = N >> 5;             // 32x32 macro tiles
  long long tilesPer = (long long)tilesM * tilesN;
  long long totalW = tilesPer * batch;
  if (wave >= totalW) return;                       // wave-uniform exit
  int b   = (int)(wave / tilesPer);
  int rem = (int)(wave % tilesPer);
  int tm  = (rem / tilesN) << 5;
  int tn  = (rem % tilesN) << 5;

  const _Float16* Ab  = A  + (long long)b * sA;
  const _Float16* Bb_ = Bt + (long long)(b / bdivB) * sB;
  float*          Cb  = C  + (long long)b * sC;

  int half = lane >> 4, l = lane & 15;
  const _Float16* ar0 = Ab  + (long long)(tm + l) * K + half * 8;
  const _Float16* ar1 = ar0 + (long long)16 * K;
  const _Float16* br0 = Bb_ + (long long)(tn + l) * K + half * 8;
  const _Float16* br1 = br0 + (long long)16 * K;

  v8f acc00 = {}, acc01 = {}, acc10 = {}, acc11 = {};
  for (int k0 = 0; k0 < K; k0 += 32) {
    v16h a0 = frag_ld(ar0, k0);
    v16h a1 = frag_ld(ar1, k0);
    v16h b0 = frag_ld(br0, k0);
    v16h b1 = frag_ld(br1, k0);
    __builtin_prefetch(ar0 + k0 + 64, 0, 0);        // global_prefetch_b8
    __builtin_prefetch(ar1 + k0 + 64, 0, 0);
    __builtin_prefetch(br0 + k0 + 64, 0, 0);
    __builtin_prefetch(br1 + k0 + 64, 0, 0);
    acc00 = __builtin_amdgcn_wmma_f32_16x16x32_f16(false, a0, false, b0, (short)0, acc00, false, false);
    acc01 = __builtin_amdgcn_wmma_f32_16x16x32_f16(false, a0, false, b1, (short)0, acc01, false, false);
    acc10 = __builtin_amdgcn_wmma_f32_16x16x32_f16(false, a1, false, b0, (short)0, acc10, false, false);
    acc11 = __builtin_amdgcn_wmma_f32_16x16x32_f16(false, a1, false, b1, (short)0, acc11, false, false);
  }
  // C layout per 16x16 tile: row = tile_m + half*8 + v, col = tile_n + l
  float* c00 = Cb + (long long)(tm + half * 8) * N + tn + l;
  float* c10 = c00 + (long long)16 * N;
#pragma unroll
  for (int v = 0; v < 8; ++v) {
    c00[(long long)v * N]      = acc00[v];
    c00[(long long)v * N + 16] = acc01[v];
    c10[(long long)v * N]      = acc10[v];
    c10[(long long)v * N + 16] = acc11[v];
  }
}

// ---------------- weight conversion (f32 -> transposed f16) ----------------
__global__ void k_cvt_wq(const float* __restrict__ wq, _Float16* __restrict__ o) {
  int i = blockIdx.x * blockDim.x + threadIdx.x;
  int total = Dd * EAv * HDv * DIMv;
  if (i >= total) return;
  int dim = i % DIMv; int r = i / DIMv;
  int h = r % HDv; r /= HDv;
  int e = r % EAv; int d = r / EAv;
  o[((long long)(d * EAv + e) * HDv + h) * DIMv + dim] =
      (_Float16)wq[(((long long)(d * EAv + e) * DIMv + dim) * HDv) + h];
}
__global__ void k_cvt_wkv(const float* __restrict__ wkv, _Float16* __restrict__ o) {
  int i = blockIdx.x * blockDim.x + threadIdx.x;
  int total = Dd * 2 * HDv * DIMv;
  if (i >= total) return;
  int dim = i % DIMv; int r = i / DIMv;
  int n = r % (2 * HDv); int d = r / (2 * HDv);
  o[((long long)d * 2 * HDv + n) * DIMv + dim] =
      (_Float16)wkv[((long long)d * DIMv + dim) * (2 * HDv) + n];
}
__global__ void k_cvt_wout(const float* __restrict__ wo, _Float16* __restrict__ o) {
  int i = blockIdx.x * blockDim.x + threadIdx.x;
  int total = Dd * DIMv * EAv * HDv;
  if (i >= total) return;
  int eh = i % (EAv * HDv); int r = i / (EAv * HDv);
  int dim = r % DIMv; int d = r / DIMv;
  int e = eh / HDv, h = eh % HDv;
  o[((long long)d * DIMv + dim) * (EAv * HDv) + eh] =
      (_Float16)wo[(((long long)(d * EAv + e) * HDv + h) * DIMv) + dim];
}
__global__ void k_cvt_w1(const float* __restrict__ w1, _Float16* __restrict__ o) {
  int i = blockIdx.x * blockDim.x + threadIdx.x;
  int total = Dd * EMv * HMv * DIMv;
  if (i >= total) return;
  int dim = i % DIMv; int r = i / DIMv;
  int hm = r % HMv; r /= HMv;
  int e = r % EMv; int d = r / EMv;
  o[((long long)(d * EMv + e) * HMv + hm) * DIMv + dim] =
      (_Float16)w1[(((long long)(d * EMv + e) * DIMv + dim) * HMv) + hm];
}
__global__ void k_cvt_w2(const float* __restrict__ w2, _Float16* __restrict__ o) {
  int i = blockIdx.x * blockDim.x + threadIdx.x;
  int total = Dd * DIMv * EMv * HMv;
  if (i >= total) return;
  int eh = i % (EMv * HMv); int r = i / (EMv * HMv);
  int dim = r % DIMv; int d = r / DIMv;
  int e = eh / HMv, hm = eh % HMv;
  o[((long long)d * DIMv + dim) * (EMv * HMv) + eh] =
      (_Float16)w2[(((long long)(d * EMv + e) * HMv + hm) * DIMv) + dim];
}

// ---------------- elementwise helpers ----------------
__global__ void k_add_te(float* __restrict__ xt, const float* __restrict__ te) {
  int i = blockIdx.x * blockDim.x + threadIdx.x;
  if (i >= TOK * DIMv) return;
  xt[i] += te[i % DIMv];
}
__global__ void k_add(float* __restrict__ a, const float* __restrict__ b, int n) {
  int i = blockIdx.x * blockDim.x + threadIdx.x;
  if (i >= n) return;
  a[i] += b[i];
}
__global__ void k_bias_add(float* __restrict__ c, const float* __restrict__ bias, int N, int n) {
  int i = blockIdx.x * blockDim.x + threadIdx.x;
  if (i >= n) return;
  c[i] += bias[i % N];
}

// ---------------- LayerNorm (per token row) ----------------
__global__ void k_ln(const float* __restrict__ x, const float* __restrict__ w,
                     const float* __restrict__ b, float* __restrict__ xf,
                     _Float16* __restrict__ xf16) {
  int t = blockIdx.x;
  const float* row = x + (long long)t * DIMv;
  __shared__ float s1[128], s2[128];
  float a = 0.f, q = 0.f;
  for (int i = threadIdx.x; i < DIMv; i += 128) { float v = row[i]; a += v; q += v * v; }
  s1[threadIdx.x] = a; s2[threadIdx.x] = q; __syncthreads();
  for (int s = 64; s > 0; s >>= 1) {
    if (threadIdx.x < s) { s1[threadIdx.x] += s1[threadIdx.x + s]; s2[threadIdx.x] += s2[threadIdx.x + s]; }
    __syncthreads();
  }
  float mean = s1[0] / DIMv;
  float var  = s2[0] / DIMv - mean * mean;
  float inv  = rsqrtf(var + 1e-5f);
  for (int i = threadIdx.x; i < DIMv; i += 128) {
    float v = (row[i] - mean) * inv * w[i] + b[i];
    xf[(long long)t * DIMv + i] = v;
    xf16[(long long)t * DIMv + i] = (_Float16)v;
  }
}

// ---------------- gating ----------------
__global__ void k_gate_logits(const float* __restrict__ xf, const float* __restrict__ wg,
                              float* __restrict__ glog, int E) {
  int i = blockIdx.x * blockDim.x + threadIdx.x;
  if (i >= TOK * E) return;
  int e = i % E, t = i / E;
  const float* xr = xf + (long long)t * DIMv;
  float s = 0.f;
  for (int k = 0; k < DIMv; ++k) s += xr[k] * wg[(long long)k * 2 * E + e];
  glog[(long long)t * E + e] = s;
}

__global__ void k_gate_topk(const float* __restrict__ glog, int E, int K,
                            int* __restrict__ topi, float* __restrict__ gateo,
                            float* __restrict__ statP, float* __restrict__ statF,
                            float* __restrict__ statZ) {
  int t = blockIdx.x * blockDim.x + threadIdx.x;
  if (t >= TOK) return;
  float lg[EAv];
  float mx = -1e30f;
  for (int e = 0; e < E; ++e) { lg[e] = glog[(long long)t * E + e]; mx = fmaxf(mx, lg[e]); }
  float sum = 0.f;
  for (int e = 0; e < E; ++e) { lg[e] = expf(lg[e] - mx); sum += lg[e]; }
  float invs = 1.f / sum;
  bool used[EAv];
  for (int e = 0; e < E; ++e) used[e] = false;
  int idx[Hh]; float gv[Hh]; float gsum = 0.f;
  for (int kk = 0; kk < K; ++kk) {
    int best = 0; float bv = -1e30f;
    for (int e = 0; e < E; ++e) {
      float p = lg[e] * invs;
      if (!used[e] && p > bv) { bv = p; best = e; }
    }
    used[best] = true; idx[kk] = best; gv[kk] = bv; gsum += bv;
  }
  float invg = 1.f / (gsum + 1e-6f);
  for (int kk = 0; kk < K; ++kk) {
    topi[(long long)t * K + kk]  = idx[kk];
    gateo[(long long)t * K + kk] = gv[kk] * invg;
    atomicAdd(&statF[idx[kk]], 1.f);
  }
  for (int e = 0; e < E; ++e) atomicAdd(&statP[e], lg[e] * invs);
  float lse = logf(sum) + mx;
  atomicAdd(&statZ[0], lse * lse);
  atomicAdd(&statZ[1], 1.f);
}

// ---------------- attention helpers ----------------
__global__ void k_gather_q16(const float* __restrict__ qall, const int* __restrict__ topi,
                             _Float16* __restrict__ q16) {
  int i = blockIdx.x * blockDim.x + threadIdx.x;
  if (i >= TOK * Hh * HDv) return;
  int dd = i % HDv; int r = i / HDv;
  int h = r % Hh; int t = r / Hh;
  int b = t / Nn, ii = t % Nn;
  int e = topi[(long long)t * Hh + h];
  q16[(((long long)(b * Hh + h)) * Nn + ii) * HDv + dd] =
      (_Float16)qall[(long long)t * (EAv * HDv) + e * HDv + dd];
}
__global__ void k_pack_kv16(const float* __restrict__ kv, _Float16* __restrict__ k16,
                            _Float16* __restrict__ v16t) {
  int i = blockIdx.x * blockDim.x + threadIdx.x;
  if (i >= TOK * HDv) return;
  int dd = i % HDv; int t = i / HDv;
  int b = t / Nn, ii = t % Nn;
  k16[((long long)b * Nn + ii) * HDv + dd]  = (_Float16)kv[(long long)t * (2 * HDv) + dd];
  v16t[((long long)b * HDv + dd) * Nn + ii] = (_Float16)kv[(long long)t * (2 * HDv) + HDv + dd];
}
__global__ void k_softmax_att(const float* __restrict__ sc, _Float16* __restrict__ att) {
  long long row = blockIdx.x;                    // 48*384 rows
  const float* r = sc + row * Nn;
  __shared__ float red[128];
  int i = threadIdx.x;
  float mx = -1e30f;
  for (int j = i; j < Nn; j += 128) mx = fmaxf(mx, r[j] * SCALEv);
  red[i] = mx; __syncthreads();
  for (int s = 64; s > 0; s >>= 1) { if (i < s) red[i] = fmaxf(red[i], red[i + s]); __syncthreads(); }
  mx = red[0]; __syncthreads();
  float sum = 0.f;
  for (int j = i; j < Nn; j += 128) sum += expf(r[j] * SCALEv - mx);
  red[i] = sum; __syncthreads();
  for (int s = 64; s > 0; s >>= 1) { if (i < s) red[i] += red[i + s]; __syncthreads(); }
  float inv = 1.f / red[0];
  for (int j = i; j < Nn; j += 128)
    att[row * Nn + j] = (_Float16)(expf(r[j] * SCALEv - mx) * inv);
}
__global__ void k_combine_attn(const float* __restrict__ o, const int* __restrict__ topi,
                               const float* __restrict__ gate, _Float16* __restrict__ c16) {
  int i = blockIdx.x * blockDim.x + threadIdx.x;
  if (i >= TOK * EAv * HDv) return;
  int dd = i % HDv; int r = i / HDv;
  int e = r % EAv; int t = r / EAv;
  int b = t / Nn, ii = t % Nn;
  float acc = 0.f;
  for (int h = 0; h < Hh; ++h) {
    if (topi[(long long)t * Hh + h] == e)
      acc += gate[(long long)t * Hh + h] *
             o[(((long long)(b * Hh + h)) * Nn + ii) * HDv + dd];
  }
  c16[(long long)t * (EAv * HDv) + e * HDv + dd] = (_Float16)acc;
}

// ---------------- MoE MLP helpers ----------------
__global__ void k_mlp_scatter(const float* __restrict__ hall, const float* __restrict__ b1,
                              const int* __restrict__ topi, const float* __restrict__ gate,
                              _Float16* __restrict__ c2) {
  long long i = (long long)blockIdx.x * blockDim.x + threadIdx.x;
  if (i >= (long long)TOK * EMv * HMv) return;
  int hm = (int)(i % HMv); long long r = i / HMv;
  int e = (int)(r % EMv); int t = (int)(r / EMv);
  float v = 0.f;
  for (int k = 0; k < KMv; ++k) {
    if (topi[(long long)t * KMv + k] == e) {
      float x = hall[(long long)t * (EMv * HMv) + e * HMv + hm] + b1[(long long)e * HMv + hm];
      float g = 0.5f * x * (1.f + erff(x * 0.70710678118654752f));
      v = gate[(long long)t * KMv + k] * g;
    }
  }
  c2[(long long)t * (EMv * HMv) + e * HMv + hm] = (_Float16)v;
}
__global__ void k_bias2_residual(float* __restrict__ xt, const float* __restrict__ y,
                                 const int* __restrict__ topi, const float* __restrict__ gate,
                                 const float* __restrict__ b2) {
  int i = blockIdx.x * blockDim.x + threadIdx.x;
  if (i >= TOK * DIMv) return;
  int n = i % DIMv; int t = i / DIMv;
  float extra = 0.f;
  for (int k = 0; k < KMv; ++k) {
    int e = topi[(long long)t * KMv + k];
    extra += gate[(long long)t * KMv + k] * b2[(long long)e * DIMv + n];
  }
  xt[i] += y[i] + extra;
}

// ---------------- classifier head ----------------
__global__ void k_head(const float* __restrict__ xt, const float* __restrict__ fcw,
                       const float* __restrict__ fcb, const float* __restrict__ hlnw,
                       const float* __restrict__ hlnb, const float* __restrict__ hw,
                       const float* __restrict__ hb, float* __restrict__ outp) {
  int b = blockIdx.x, i = threadIdx.x;           // 384 threads
  __shared__ float r1[DIMv], r2[DIMv];
  float s = 0.f;
  for (int j = 1; j < Nn; ++j) s += xt[((long long)(b * Nn + j)) * DIMv + i];
  s /= (float)(Nn - 1);
  r1[i] = s; r2[i] = s * s; __syncthreads();
  for (int st = 256; st > 0; st >>= 1) {
    if (i < st && i + st < DIMv) { r1[i] += r1[i + st]; r2[i] += r2[i + st]; }
    __syncthreads();
  }
  float mean = r1[0] / DIMv, var = r2[0] / DIMv - mean * mean;
  float v1 = (s - mean) * rsqrtf(var + 1e-5f) * fcw[i] + fcb[i];
  __syncthreads();
  r1[i] = v1; r2[i] = v1 * v1; __syncthreads();
  for (int st = 256; st > 0; st >>= 1) {
    if (i < st && i + st < DIMv) { r1[i] += r1[i + st]; r2[i] += r2[i + st]; }
    __syncthreads();
  }
  mean = r1[0] / DIMv; var = r2[0] / DIMv - mean * mean;
  float v2 = (v1 - mean) * rsqrtf(var + 1e-5f) * hlnw[i] + hlnb[i];
  __syncthreads();
  r1[i] = v2 * hw[(long long)i * NCLSv + 0];
  r2[i] = v2 * hw[(long long)i * NCLSv + 1];
  __syncthreads();
  for (int st = 256; st > 0; st >>= 1) {
    if (i < st && i + st < DIMv) { r1[i] += r1[i + st]; r2[i] += r2[i + st]; }
    __syncthreads();
  }
  if (i == 0) {
    outp[b * NCLSv + 0] = r1[0] + hb[0];
    outp[b * NCLSv + 1] = r2[0] + hb[1];
  }
}

// ---------------- aux loss ----------------
// stats layout per layer d (44 floats): pa[12] fa[12] za[2] pm[8] fm[8] zm[2]
__global__ void k_aux(const float* __restrict__ stats, float* __restrict__ zout) {
  float z = 0.f;
  for (int d = 0; d < Dd; ++d) {
    const float* s = stats + d * 44;
    float ps = 0.f, fs = 0.f;
    for (int e = 0; e < EAv; ++e) { ps += s[e]; fs += s[12 + e]; }
    float sw = 0.f;
    for (int e = 0; e < EAv; ++e) sw += (s[e] / ps) * (s[12 + e] / fs);
    z += SWITCHv * (float)EAv * sw + ZLOSSv * s[24] / s[25];
    const float* m = s + 26;
    ps = 0.f; fs = 0.f;
    for (int e = 0; e < EMv; ++e) { ps += m[e]; fs += m[8 + e]; }
    sw = 0.f;
    for (int e = 0; e < EMv; ++e) sw += (m[e] / ps) * (m[8 + e] / fs);
    z += SWITCHv * (float)EMv * sw + ZLOSSv * m[16] / m[17];
  }
  zout[0] = z;
}

// =====================================================================
// host orchestration
// =====================================================================
static inline void launch_gemm(const _Float16* A, const _Float16* Bt, float* C,
                               int M, int N, int K, int batch, long long sA,
                               long long sB, int bdivB, long long sC,
                               hipStream_t st) {
  long long waves = (long long)(M / 32) * (N / 32) * batch;  // 32x32 per wave
  long long threads = waves * 32;
  unsigned blocks = (unsigned)((threads + 255) / 256);
  gemm_wmma_f16<<<blocks, 256, 0, st>>>(A, Bt, C, M, N, K, sA, sB, bdivB, sC, batch);
}

#define EW_GRID(n) dim3((unsigned)(((n) + 255) / 256)), dim3(256), 0, stream

extern "C" void kernel_launch(void* const* d_in, const int* in_sizes, int n_in,
                              void* d_out, int out_size, void* d_ws, size_t ws_size,
                              hipStream_t stream) {
  (void)in_sizes; (void)n_in; (void)out_size; (void)ws_size;

  const float* x       = (const float*)d_in[0];
  const float* te      = (const float*)d_in[1];
  const float* ln1w    = (const float*)d_in[2];
  const float* ln1b    = (const float*)d_in[3];
  const float* wg_attn = (const float*)d_in[4];
  const float* wq      = (const float*)d_in[5];
  const float* wout    = (const float*)d_in[6];
  const float* wkv     = (const float*)d_in[7];
  const float* bkv     = (const float*)d_in[8];
  const float* ln2w    = (const float*)d_in[9];
  const float* ln2b    = (const float*)d_in[10];
  const float* wg_mlp  = (const float*)d_in[11];
  const float* w1      = (const float*)d_in[12];
  const float* b1      = (const float*)d_in[13];
  const float* w2      = (const float*)d_in[14];
  const float* b2      = (const float*)d_in[15];
  const float* fcw     = (const float*)d_in[16];
  const float* fcb     = (const float*)d_in[17];
  const float* hlnw    = (const float*)d_in[18];
  const float* hlnb    = (const float*)d_in[19];
  const float* hw      = (const float*)d_in[20];
  const float* hb      = (const float*)d_in[21];
  float* outp = (float*)d_out;

  // ---------------- workspace bump allocator ----------------
  char* wsp = (char*)d_ws;
  auto alloc = [&](size_t bytes) -> void* {
    void* p = (void*)wsp;
    wsp += (bytes + 255) & ~(size_t)255;
    return p;
  };
  _Float16* wq16   = (_Float16*)alloc((size_t)Dd * EAv * HDv * DIMv * 2);
  _Float16* wkv16  = (_Float16*)alloc((size_t)Dd * 2 * HDv * DIMv * 2);
  _Float16* wout16 = (_Float16*)alloc((size_t)Dd * DIMv * EAv * HDv * 2);
  _Float16* w1t16  = (_Float16*)alloc((size_t)Dd * EMv * HMv * DIMv * 2);
  _Float16* w2t16  = (_Float16*)alloc((size_t)Dd * DIMv * EMv * HMv * 2);
  float*    xt     = (float*)alloc((size_t)TOK * DIMv * 4);
  float*    xf     = (float*)alloc((size_t)TOK * DIMv * 4);
  _Float16* xf16   = (_Float16*)alloc((size_t)TOK * DIMv * 2);
  float*    glog   = (float*)alloc((size_t)TOK * EAv * 4);
  int*      topiA  = (int*)alloc((size_t)TOK * Hh * 4);
  float*    gateA  = (float*)alloc((size_t)TOK * Hh * 4);
  int*      topiM  = (int*)alloc((size_t)TOK * KMv * 4);
  float*    gateM  = (float*)alloc((size_t)TOK * KMv * 4);
  float*    qall   = (float*)alloc((size_t)TOK * EAv * HDv * 4);
  float*    kvb    = (float*)alloc((size_t)TOK * 2 * HDv * 4);
  _Float16* q16    = (_Float16*)alloc((size_t)Bb * Hh * Nn * HDv * 2);
  _Float16* k16    = (_Float16*)alloc((size_t)Bb * Nn * HDv * 2);
  _Float16* v16t   = (_Float16*)alloc((size_t)Bb * HDv * Nn * 2);
  float*    big0   = (float*)alloc((size_t)TOK * EMv * HMv * 4);     // scores / h_all
  _Float16* big1   = (_Float16*)alloc((size_t)TOK * EMv * HMv * 2);  // att16 / c2
  float*    ob     = (float*)alloc((size_t)Bb * Hh * Nn * HDv * 4);
  _Float16* c16    = (_Float16*)alloc((size_t)TOK * EAv * HDv * 2);
  float*    yb     = (float*)alloc((size_t)TOK * DIMv * 4);
  float*    stats  = (float*)alloc((size_t)Dd * 44 * 4);

  // ---------------- one-time per-launch weight conversion ----------------
  k_cvt_wq  <<<EW_GRID(Dd * EAv * HDv * DIMv)>>>(wq, wq16);
  k_cvt_wkv <<<EW_GRID(Dd * 2 * HDv * DIMv)>>>(wkv, wkv16);
  k_cvt_wout<<<EW_GRID(Dd * DIMv * EAv * HDv)>>>(wout, wout16);
  k_cvt_w1  <<<EW_GRID((int)((size_t)Dd * EMv * HMv * DIMv))>>>(w1, w1t16);
  k_cvt_w2  <<<EW_GRID((int)((size_t)Dd * DIMv * EMv * HMv))>>>(w2, w2t16);
  hipMemsetAsync(stats, 0, (size_t)Dd * 44 * 4, stream);

  for (int t = 0; t < Tt; ++t) {
    hipMemcpyAsync(xt, x, (size_t)TOK * DIMv * 4, hipMemcpyDeviceToDevice, stream);
    for (int d = 0; d < Dd; ++d) {
      float* sA = stats + d * 44;        // pa fa za
      float* sM = stats + d * 44 + 26;   // pm fm zm

      // x += task_embedding[t]; LN1
      k_add_te<<<EW_GRID(TOK * DIMv)>>>(xt, te + (long long)t * DIMv);
      k_ln<<<dim3(TOK), dim3(128), 0, stream>>>(xt, ln1w + d * DIMv, ln1b + d * DIMv, xf, xf16);

      // attention gate (top-6 of 12) + stats
      k_gate_logits<<<EW_GRID(TOK * EAv)>>>(xf, wg_attn + ((long long)(d * Tt + t)) * DIMv * 2 * EAv, glog, EAv);
      k_gate_topk<<<EW_GRID(TOK)>>>(glog, EAv, Hh, topiA, gateA, sA, sA + 12, sA + 24);

      // q_all = xf @ Wq (all 12 experts), kv = xf @ Wkv + bkv
      launch_gemm(xf16, wq16 + (long long)d * EAv * HDv * DIMv, qall,
                  TOK, EAv * HDv, DIMv, 1, 0, 0, 1, 0, stream);
      launch_gemm(xf16, wkv16 + (long long)d * 2 * HDv * DIMv, kvb,
                  TOK, 2 * HDv, DIMv, 1, 0, 0, 1, 0, stream);
      k_bias_add<<<EW_GRID(TOK * 2 * HDv)>>>(kvb, bkv + d * 2 * HDv, 2 * HDv, TOK * 2 * HDv);

      // gather top-k expert queries; pack K and V^T as f16
      k_gather_q16<<<EW_GRID(TOK * Hh * HDv)>>>(qall, topiA, q16);
      k_pack_kv16<<<EW_GRID(TOK * HDv)>>>(kvb, k16, v16t);

      // scores[bh] = Q[bh] @ K[b]^T  (48 batched 384x384x64)
      launch_gemm(q16, k16, big0, Nn, Nn, HDv, Bb * Hh,
                  (long long)Nn * HDv, (long long)Nn * HDv, Hh, (long long)Nn * Nn, stream);
      k_softmax_att<<<dim3(Bb * Hh * Nn), dim3(128), 0, stream>>>(big0, big1);
      // o[bh] = att[bh] @ V[b]  (48 batched 384x64x384)
      launch_gemm(big1, v16t, ob, Nn, HDv, Nn, Bb * Hh,
                  (long long)Nn * Nn, (long long)HDv * Nn, Hh, (long long)Nn * HDv, stream);

      // expert combine + output projection; residual
      k_combine_attn<<<EW_GRID(TOK * EAv * HDv)>>>(ob, topiA, gateA, c16);
      launch_gemm(c16, wout16 + (long long)d * DIMv * EAv * HDv, yb,
                  TOK, DIMv, EAv * HDv, 1, 0, 0, 1, 0, stream);
      k_add<<<EW_GRID(TOK * DIMv)>>>(xt, yb, TOK * DIMv);

      // ---- MoE MLP ----
      k_ln<<<dim3(TOK), dim3(128), 0, stream>>>(xt, ln2w + d * DIMv, ln2b + d * DIMv, xf, xf16);
      k_gate_logits<<<EW_GRID(TOK * EMv)>>>(xf, wg_mlp + ((long long)(d * Tt + t)) * DIMv * 2 * EMv, glog, EMv);
      k_gate_topk<<<EW_GRID(TOK)>>>(glog, EMv, KMv, topiM, gateM, sM, sM + 8, sM + 16);

      // h_all = xf @ W1 (all 8 experts); gelu+gate scatter; out = c2 @ W2
      launch_gemm(xf16, w1t16 + (long long)d * EMv * HMv * DIMv, big0,
                  TOK, EMv * HMv, DIMv, 1, 0, 0, 1, 0, stream);
      {
        long long n = (long long)TOK * EMv * HMv;
        k_mlp_scatter<<<dim3((unsigned)((n + 255) / 256)), dim3(256), 0, stream>>>(
            big0, b1 + (long long)d * EMv * HMv, topiM, gateM, big1);
      }
      launch_gemm(big1, w2t16 + (long long)d * DIMv * EMv * HMv, yb,
                  TOK, DIMv, EMv * HMv, 1, 0, 0, 1, 0, stream);
      k_bias2_residual<<<EW_GRID(TOK * DIMv)>>>(xt, yb, topiM, gateM, b2 + (long long)d * EMv * DIMv);
    }
    // classifier head for this task
    k_head<<<dim3(Bb), dim3(DIMv), 0, stream>>>(
        xt, fcw, fcb, hlnw + t * DIMv, hlnb + t * DIMv,
        hw + (long long)t * DIMv * NCLSv, hb + t * NCLSv,
        outp + (long long)t * Bb * NCLSv);
  }
  // aux loss scalar (last output element)
  k_aux<<<dim3(1), dim3(1), 0, stream>>>(stats, outp + (long long)Tt * Bb * NCLSv);
}
